// DMRGCNGPGraph_85177791415025
// MI455X (gfx1250) — matrix operand order repaired and checked by reading, not compiled, not run on hardware
//
#include <hip/hip_runtime.h>

// ---------------------------------------------------------------------------
// DMRGCN forward for MI455X (gfx1250, wave32, WMMA).
//
// Input flatten order assumed (per "setup_inputs() dict order"):
//   d_in[0]  = V_obs  (8,8,256,2) f32
//   d_in[1]  = A_obs  (8,3,8,256,256) f32
//   d_in[2..6]   = w[0..4]   (R,cin,cout) f32
//   d_in[7..11]  = b[0..4]   (cout,) f32
//   d_in[12..15] = tk[0..3]  (cout,cout,3,1) f32
//   d_in[16] = wt (8,12) f32
// d_out = pred (8,5,12,256) f32  ++  gi (8,256) as float
// Workspace use: ~38 MB.
//
// Perf notes: bf16 WMMA (16x16x32) everywhere; adjacency stored transposed in
// bf16 so every WMMA operand is 2x b128 contiguous loads; dual accumulators
// per wave so independent WMMAs fill the bf16 WMMA hazard slots instead of
// v_nops, and the shared A-operand is loaded once per tile pair.
// ---------------------------------------------------------------------------

typedef __attribute__((ext_vector_type(16))) __bf16 v16bf;
typedef __attribute__((ext_vector_type(8)))  float  v8f;

struct alignas(16) I4 { int x, y, z, w; };
struct alignas(16) F4 { float x, y, z, w; };

__device__ __forceinline__ unsigned short f2bf(float f) {
  unsigned int u = __float_as_uint(f);
  u += 0x7fffu + ((u >> 16) & 1u);           // round-to-nearest-even
  return (unsigned short)(u >> 16);
}

union VU { v16bf v; I4 q[2]; };

__device__ __forceinline__ v16bf ld2(const unsigned short* p0, const unsigned short* p1) {
  VU u;
  u.q[0] = *reinterpret_cast<const I4*>(p0);
  u.q[1] = *reinterpret_cast<const I4*>(p1);
  return u.v;
}
__device__ __forceinline__ v16bf bzero16() {
  VU u; u.q[0] = I4{0,0,0,0}; u.q[1] = I4{0,0,0,0}; return u.v;
}
__device__ __forceinline__ I4 pack8(const float* f) {
  unsigned r0 = (unsigned)f2bf(f[0]) | ((unsigned)f2bf(f[1]) << 16);
  unsigned r1 = (unsigned)f2bf(f[2]) | ((unsigned)f2bf(f[3]) << 16);
  unsigned r2 = (unsigned)f2bf(f[4]) | ((unsigned)f2bf(f[5]) << 16);
  unsigned r3 = (unsigned)f2bf(f[6]) | ((unsigned)f2bf(f[7]) << 16);
  return I4{(int)r0, (int)r1, (int)r2, (int)r3};
}

#define WMMA_BF16(A, B, C) \
  __builtin_amdgcn_wmma_f32_16x16x32_bf16(false, (A), false, (B), (short)0, (C), false, false)

// ---------------------------------------------------------------------------
// Group assignment: threshold graph on last positions + 16 min-label sweeps.
// ---------------------------------------------------------------------------
__global__ void k_group(const float* __restrict__ V, int* __restrict__ gi,
                        float* __restrict__ gi_out) {
  __shared__ float lx[256], ly[256];
  __shared__ int lab[256];
  const int b = blockIdx.x, n = threadIdx.x;
  const float* p = V + (((size_t)b * 8 + 7) * 256 + n) * 2;
  lx[n] = p[0]; ly[n] = p[1]; lab[n] = n;
  __syncthreads();
  for (int it = 0; it < 16; ++it) {
    int m = lab[n];
    const float x = lx[n], y = ly[n];
    for (int j = 0; j < 256; ++j) {
      float dx = x - lx[j], dy = y - ly[j];
      if (dx * dx + dy * dy <= 4.0f) m = min(m, lab[j]);
    }
    __syncthreads();
    lab[n] = m;
    __syncthreads();
  }
  gi[b * 256 + n] = lab[n];
  gi_out[b * 256 + n] = (float)lab[n];
}

// ---------------------------------------------------------------------------
// Prep: bf16 weight repack (Wt[l][r][d][c64], Km[l][d][dt*64+c]), X0 pad.
// ---------------------------------------------------------------------------
__global__ void k_prep(const float* V,
                       const float* w0, const float* w1, const float* w2,
                       const float* w3, const float* w4,
                       const float* k0, const float* k1, const float* k2, const float* k3,
                       unsigned short* Wt, unsigned short* Km, unsigned short* X0) {
  const int idx = blockIdx.x * 256 + threadIdx.x;
  const int NWT = 5 * 3 * 64 * 64;    // 61440
  const int NKM = 4 * 64 * 192;       // 49152
  const int NX0 = 8 * 8 * 256 * 64;   // 1048576
  if (idx < NWT) {
    int l = idx / (3 * 64 * 64);
    int rem = idx % (3 * 64 * 64);
    int r = rem / (64 * 64);
    int d = (rem / 64) % 64;
    int c = rem % 64;
    const int cin[5] = {2, 64, 64, 64, 64};
    const int cout[5] = {64, 64, 64, 64, 5};
    const float* wp = (l == 0) ? w0 : (l == 1) ? w1 : (l == 2) ? w2 : (l == 3) ? w3 : w4;
    float v = 0.0f;
    if (c < cin[l] && d < cout[l]) v = wp[(r * cin[l] + c) * cout[l] + d];
    Wt[idx] = f2bf(v);
  } else if (idx < NWT + NKM) {
    int j = idx - NWT;
    int l = j / (64 * 192);
    int d = (j / 192) % 64;
    int k = j % 192;
    int dt = k / 64, c = k % 64;
    const float* kp = (l == 0) ? k0 : (l == 1) ? k1 : (l == 2) ? k2 : k3;
    Km[j] = f2bf(kp[(d * 64 + c) * 3 + dt]);
  } else if (idx < NWT + NKM + NX0) {
    int j = idx - NWT - NKM;
    int c = j & 63;
    int n = (j >> 6) & 255;
    int bt = j >> 14;
    float v = (c < 2) ? V[(((size_t)bt) * 256 + n) * 2 + c] : 0.0f;
    X0[j] = f2bf(v);
  }
}

// ---------------------------------------------------------------------------
// Adjacency builder: writes bf16 TRANSPOSED adjacency adjT[b][r][t][m][n].
// family 0: A_agent (scale-thresholded A_obs[:,1])
// family 1: A_intra (same-group mask)
// family 2: A_inter (group-pair connectivity via LDS bitmap)
// ---------------------------------------------------------------------------
__global__ void k_build_adj(const float* __restrict__ A, const int* __restrict__ gi,
                            unsigned short* __restrict__ adjT, int family) {
  __shared__ unsigned short tile[64 * 65];
  __shared__ unsigned int bits[2048];
  __shared__ int gl[256];
  const int tid = threadIdx.x;
  const int b = blockIdx.x / 24;
  const int r = (blockIdx.x / 8) % 3;
  const int t = blockIdx.x % 8;
  gl[tid] = gi[b * 256 + tid];
  __syncthreads();

  const int srcR = (family == 0) ? 1 : r;   // A_agent uses A_obs[:,1]
  const float* src = A + (((size_t)b * 3 + srcR) * 8 + t) * 65536;
  unsigned short* dst = adjT + (((size_t)b * 3 + r) * 8 + t) * 65536;

  if (family == 2) {
    for (int i = tid; i < 2048; i += 256) bits[i] = 0;
    __syncthreads();
    const int gm = gl[tid];
    for (int n = 0; n < 256; ++n) {
      float v = src[(size_t)n * 256 + tid];
      int gn = gl[n];
      if (v > 0.0f && gn != gm) {
        int pidx = gn * 256 + gm;
        atomicOr(&bits[pidx >> 5], 1u << (pidx & 31));
      }
    }
    __syncthreads();
    const unsigned short one = f2bf(1.0f);
    const int gn = gl[tid];                  // tid == n in write phase
    for (int m = 0; m < 256; ++m) {
      int pidx = gn * 256 + gl[m];
      unsigned bit = (bits[pidx >> 5] >> (pidx & 31)) & 1u;
      dst[(size_t)m * 256 + tid] = bit ? one : (unsigned short)0;
    }
    return;
  }

  const float scale = (r == 0) ? 0.5f : (r == 1) ? 1.0f : 2.0f;
  const int tx = tid & 63, ty = tid >> 6;    // 64 x 4 thread tile
  for (int tb = 0; tb < 16; ++tb) {
    const int n0 = (tb >> 2) * 64, m0 = (tb & 3) * 64;
    __syncthreads();
#pragma unroll
    for (int i = 0; i < 16; ++i) {
      int row = ty + i * 4;                  // n-local
      int n = n0 + row, m = m0 + tx;
      float v = src[(size_t)n * 256 + m];
      if (family == 0) v = (v <= scale) ? v : 0.0f;
      else             v = (gl[n] == gl[m]) ? v : 0.0f;
      tile[tx * 65 + row] = f2bf(v);         // [m-local][n-local]
    }
    __syncthreads();
#pragma unroll
    for (int i = 0; i < 16; ++i) {
      int row = ty + i * 4;                  // m-local
      dst[(size_t)(m0 + row) * 256 + n0 + tx] = tile[row * 65 + tx];
    }
  }
}

// ---------------------------------------------------------------------------
// Graph conv layer (WMMA). Phase 1: h[r][d][n] = Wt[r]^T x  -> LDS (bf16).
// Phase 2: out[d][m] = sum_r h_r @ A_r  (+bias, optional ReLU).
// Both phases process two tiles per wave iteration with independent
// accumulators so interleaved WMMAs cover the bf16 WMMA hazard window.
// Output: bf16 activations [bt][n][c64]  OR  fp32 out5 [bt][n][16].
// ---------------------------------------------------------------------------
__global__ __launch_bounds__(256) void k_gconv(
    const unsigned short* __restrict__ x, const unsigned short* __restrict__ Wt,
    const unsigned short* __restrict__ adjT, const float* __restrict__ bias,
    unsigned short* __restrict__ xout, float* __restrict__ fout,
    int Dtiles, int relu, int doutReal) {
  extern __shared__ unsigned short hsh[];    // [3][Dtiles*16][256] bf16
  const int bt = blockIdx.x;
  const int tid = threadIdx.x, lane = tid & 31, wv = tid >> 5;
  const int l15 = lane & 15, hi = lane >> 4;
  const int Dp = Dtiles * 16;

  // ---- Phase 1: h = Wt^T x via WMMA (M = n rows, N = d cols, K = c) ----
  const int nht = 3 * 16 * Dtiles;           // 192 or 48 (always even)
  for (int tp = wv; tp < (nht >> 1); tp += 8) {
    const int t0 = tp * 2, t1 = t0 + 1;
    const int r0 = t0 / (16 * Dtiles), r1 = t1 / (16 * Dtiles);
    const int re0 = t0 % (16 * Dtiles), re1 = t1 % (16 * Dtiles);
    const int nt0 = re0 / Dtiles, dt0 = re0 % Dtiles;
    const int nt1 = re1 / Dtiles, dt1 = re1 % Dtiles;
    v8f acc0 = {}, acc1 = {};
    const unsigned short* xr0 = x + ((size_t)bt * 256 + nt0 * 16 + l15) * 64;
    const unsigned short* xr1 = x + ((size_t)bt * 256 + nt1 * 16 + l15) * 64;
    const unsigned short* wr0 = Wt + ((size_t)r0 * 64 + dt0 * 16 + l15) * 64;
    const unsigned short* wr1 = Wt + ((size_t)r1 * 64 + dt1 * 16 + l15) * 64;
#pragma unroll
    for (int kc = 0; kc < 2; ++kc) {
      const int klo = kc * 32 + hi * 16;
      const int kb = kc * 32 + hi * 16;
      v16bf a0 = ld2(xr0 + klo, xr0 + klo + 8);
      v16bf b0 = ld2(wr0 + kb, wr0 + kb + 8);
      v16bf a1 = ld2(xr1 + klo, xr1 + klo + 8);
      v16bf b1 = ld2(wr1 + kb, wr1 + kb + 8);
      acc0 = WMMA_BF16(a0, b0, acc0);
      acc1 = WMMA_BF16(a1, b1, acc1);
    }
    float f0[8], f1[8];
#pragma unroll
    for (int v = 0; v < 8; ++v) { f0[v] = acc0[v]; f1[v] = acc1[v]; }
    *reinterpret_cast<I4*>(
        &hsh[(size_t)(r0 * Dp + dt0 * 16 + l15) * 256 + nt0 * 16 + hi * 8]) = pack8(f0);
    *reinterpret_cast<I4*>(
        &hsh[(size_t)(r1 * Dp + dt1 * 16 + l15) * 256 + nt1 * 16 + hi * 8]) = pack8(f1);
  }
  __syncthreads();

  // ---- Phase 2: out = sum_r h_r @ A_r (M = d rows, N = m cols, K = n) ----
  const int b = bt >> 3, t = bt & 7;
  const int nout = Dtiles * 16;              // 64 or 16 (even)
  for (int tp = wv; tp < (nout >> 1); tp += 8) {
    const int tile0 = tp * 2;
    const int dtile = tile0 >> 4;            // pair shares dtile
    const int mt0 = tile0 & 15, mt1 = mt0 + 1;
    v8f acc0 = {}, acc1 = {};
    const int d = dtile * 16 + l15;
    const int m0 = mt0 * 16 + l15;
    const int m1 = mt1 * 16 + l15;
    for (int r = 0; r < 3; ++r) {
      const unsigned short* hrow = &hsh[(size_t)(r * Dp + d) * 256];
      const unsigned short* abase = adjT + (((size_t)b * 3 + r) * 8 + t) * 65536;
      const unsigned short* ar0 = abase + (size_t)m0 * 256;
      const unsigned short* ar1 = abase + (size_t)m1 * 256;
      __builtin_prefetch(ar1 + 256, 0, 1);   // next rows of the adjacency stream
#pragma unroll
      for (int kc = 0; kc < 8; ++kc) {
        const int klo = kc * 32 + hi * 16;
        const int kb = kc * 32 + hi * 16;
        v16bf a = ld2(hrow + klo, hrow + klo + 8);          // shared A operand
        v16bf b0 = ld2(ar0 + kb, ar0 + kb + 8);
        v16bf b1 = ld2(ar1 + kb, ar1 + kb + 8);
        acc0 = WMMA_BF16(a, b0, acc0);
        acc1 = WMMA_BF16(a, b1, acc1);       // independent: fills hazard slots
      }
    }
    float f0[8], f1[8];
#pragma unroll
    for (int v = 0; v < 8; ++v) {
      const int dd = dtile * 16 + hi * 8 + v;
      float bv = (dd < doutReal) ? bias[dd] : 0.0f;
      float t0v = acc0[v] + bv;
      float t1v = acc1[v] + bv;
      if (relu) { t0v = t0v > 0.0f ? t0v : 0.0f; t1v = t1v > 0.0f ? t1v : 0.0f; }
      f0[v] = t0v; f1[v] = t1v;
    }
    const int d0 = dtile * 16 + hi * 8;
    if (fout) {
      float* o0 = &fout[((size_t)bt * 256 + m0) * 16 + d0];
      float* o1 = &fout[((size_t)bt * 256 + m1) * 16 + d0];
      *reinterpret_cast<F4*>(o0)     = F4{f0[0], f0[1], f0[2], f0[3]};
      *reinterpret_cast<F4*>(o0 + 4) = F4{f0[4], f0[5], f0[6], f0[7]};
      *reinterpret_cast<F4*>(o1)     = F4{f1[0], f1[1], f1[2], f1[3]};
      *reinterpret_cast<F4*>(o1 + 4) = F4{f1[4], f1[5], f1[6], f1[7]};
    } else {
      *reinterpret_cast<I4*>(&xout[((size_t)bt * 256 + m0) * 64 + d0]) = pack8(f0);
      *reinterpret_cast<I4*>(&xout[((size_t)bt * 256 + m1) * 64 + d0]) = pack8(f1);
    }
  }
}

// ---------------------------------------------------------------------------
// Temporal conv (3,1) SAME as WMMA: out(64x256) = Km(64x192) @ Xg(192x256),
// Xg[k=dt*64+c][n] = x[b][t-1+dt][n][c] (zero pad at T edges).
// Two n-tiles per wave iteration share the Km A-operand.
// ---------------------------------------------------------------------------
__global__ __launch_bounds__(256) void k_tconv(const unsigned short* __restrict__ x,
                                               const unsigned short* __restrict__ Km,
                                               unsigned short* __restrict__ xout) {
  const int bt = blockIdx.x;
  const int b = bt >> 3, t = bt & 7;
  const int tid = threadIdx.x, lane = tid & 31, wv = tid >> 5;
  const int l15 = lane & 15, hi = lane >> 4;
  for (int tp = wv; tp < 32; tp += 8) {
    const int tile0 = tp * 2;
    const int dtile = tile0 >> 4;            // pair shares dtile
    const int nt0 = tile0 & 15, nt1 = nt0 + 1;
    v8f acc0 = {}, acc1 = {};
    const int d = dtile * 16 + l15;
    const int n0 = nt0 * 16 + l15;
    const int n1 = nt1 * 16 + l15;
    const unsigned short* krow = Km + (size_t)d * 192;
#pragma unroll
    for (int kc = 0; kc < 6; ++kc) {
      const int dt = kc >> 1;
      const int tt = t - 1 + dt;
      const int klo = kc * 32 + hi * 16;
      v16bf a = ld2(krow + klo, krow + klo + 8);            // shared A operand
      v16bf b0, b1;
      if (tt >= 0 && tt < 8) {
        const size_t roff = ((size_t)b * 8 + tt) * 256;
        const int cb = (kc & 1) * 32 + hi * 16;
        const unsigned short* xr0 = x + (roff + n0) * 64 + cb;
        const unsigned short* xr1 = x + (roff + n1) * 64 + cb;
        b0 = ld2(xr0, xr0 + 8);
        b1 = ld2(xr1, xr1 + 8);
      } else {
        b0 = bzero16();
        b1 = bzero16();
      }
      acc0 = WMMA_BF16(a, b0, acc0);
      acc1 = WMMA_BF16(a, b1, acc1);
    }
    float f0[8], f1[8];
#pragma unroll
    for (int v = 0; v < 8; ++v) { f0[v] = acc0[v]; f1[v] = acc1[v]; }
    const int d0 = dtile * 16 + hi * 8;
    *reinterpret_cast<I4*>(&xout[((size_t)bt * 256 + n0) * 64 + d0]) = pack8(f0);
    *reinterpret_cast<I4*>(&xout[((size_t)bt * 256 + n1) * 64 + d0]) = pack8(f1);
  }
}

// ---------------------------------------------------------------------------
// Final: pred[b][c][p][n] = (1/3) sum_f sum_t out5[f][bt][n][c] * wt[t][p]
// ---------------------------------------------------------------------------
__global__ void k_final(const float* __restrict__ out5, const float* __restrict__ wt,
                        float* __restrict__ pred) {
  const int idx = blockIdx.x * 256 + threadIdx.x;
  if (idx >= 8 * 5 * 12 * 256) return;
  const int n = idx & 255;
  const int p = (idx >> 8) % 12;
  const int c = ((idx >> 8) / 12) % 5;
  const int b = (idx >> 8) / 60;
  float s = 0.0f;
  for (int t = 0; t < 8; ++t) {
    const size_t o = (((size_t)b * 8 + t) * 256 + n) * 16 + c;
    const float fv = (out5[o] + out5[o + 524288] + out5[o + 2 * 524288]) * (1.0f / 3.0f);
    s += fv * wt[t * 12 + p];
  }
  pred[idx] = s;
}

// ---------------------------------------------------------------------------
extern "C" void kernel_launch(void* const* d_in, const int* in_sizes, int n_in,
                              void* d_out, int out_size, void* d_ws, size_t ws_size,
                              hipStream_t stream) {
  (void)in_sizes; (void)n_in; (void)out_size; (void)ws_size;
  const float* V    = (const float*)d_in[0];
  const float* Aobs = (const float*)d_in[1];
  const float* w_[5]  = {(const float*)d_in[2], (const float*)d_in[3], (const float*)d_in[4],
                         (const float*)d_in[5], (const float*)d_in[6]};
  const float* b_[5]  = {(const float*)d_in[7], (const float*)d_in[8], (const float*)d_in[9],
                         (const float*)d_in[10], (const float*)d_in[11]};
  const float* tk_[4] = {(const float*)d_in[12], (const float*)d_in[13],
                         (const float*)d_in[14], (const float*)d_in[15]};
  const float* wt = (const float*)d_in[16];
  float* out = (float*)d_out;

  char* ws = (char*)d_ws;
  int*            gi   = (int*)(ws + 0);
  unsigned short* X0   = (unsigned short*)(ws + 8192);
  unsigned short* xa   = (unsigned short*)(ws + 2105344);
  unsigned short* xb   = (unsigned short*)(ws + 4202496);
  unsigned short* Wt   = (unsigned short*)(ws + 6299648);
  unsigned short* Km   = (unsigned short*)(ws + 6422528);
  float*          out5 = (float*)(ws + 6520832);
  unsigned short* adjT = (unsigned short*)(ws + 12812288);   // end ~38 MB

  k_group<<<8, 256, 0, stream>>>(V, gi, out + 122880);
  k_prep<<<4528, 256, 0, stream>>>(V, w_[0], w_[1], w_[2], w_[3], w_[4],
                                   tk_[0], tk_[1], tk_[2], tk_[3], Wt, Km, X0);

  const size_t HBIG = 3 * 64 * 256 * 2;   // 98304 B LDS for Dout=64
  const size_t HSML = 3 * 16 * 256 * 2;   // 24576 B LDS for Dout=5 (padded 16)
  for (int f = 0; f < 3; ++f) {
    k_build_adj<<<192, 256, 0, stream>>>(Aobs, gi, adjT, f);
    // layer 0..3: gconv(+bias+ReLU) then temporal conv
    k_gconv<<<64, 256, HBIG, stream>>>(X0, Wt + 0 * 12288, adjT, b_[0], xa, nullptr, 4, 1, 64);
    k_tconv<<<64, 256, 0, stream>>>(xa, Km + 0 * 12288, xb);
    k_gconv<<<64, 256, HBIG, stream>>>(xb, Wt + 1 * 12288, adjT, b_[1], xa, nullptr, 4, 1, 64);
    k_tconv<<<64, 256, 0, stream>>>(xa, Km + 1 * 12288, xb);
    k_gconv<<<64, 256, HBIG, stream>>>(xb, Wt + 2 * 12288, adjT, b_[2], xa, nullptr, 4, 1, 64);
    k_tconv<<<64, 256, 0, stream>>>(xa, Km + 2 * 12288, xb);
    k_gconv<<<64, 256, HBIG, stream>>>(xb, Wt + 3 * 12288, adjT, b_[3], xa, nullptr, 4, 1, 64);
    k_tconv<<<64, 256, 0, stream>>>(xa, Km + 3 * 12288, xb);
    // layer 4: Dout=5 (padded to 16), no ReLU, fp32 output per family
    k_gconv<<<64, 256, HSML, stream>>>(xb, Wt + 4 * 12288, adjT, b_[4], nullptr,
                                       out5 + (size_t)f * 524288, 1, 0, 5);
  }
  k_final<<<480, 256, 0, stream>>>(out5, wt, out);
}
